// MultiHeadAttentionDec_13743895347281
// MI455X (gfx1250) — matrix operand order repaired and checked by reading, not compiled, not run on hardware
//
#include <hip/hip_runtime.h>
#include <hip/hip_bf16.h>
#include <math.h>

// Problem constants (match reference)
#define B_    256
#define M_    196
#define E_    1024
#define H_    8
#define Dh_   128
#define MID_  64
#define ALPHA_ 1.3f
#define EPS_   1e-5f

typedef __attribute__((ext_vector_type(16))) __bf16 v16bf;
typedef __attribute__((ext_vector_type(8)))  float  v8f;

union Frag16 { v16bf v; unsigned int d[8]; };

__device__ __forceinline__ unsigned short f32_bf16(float f) {
  unsigned int u = __float_as_uint(f);
  u += 0x7FFFu + ((u >> 16) & 1u);   // round-to-nearest-even
  return (unsigned short)(u >> 16);
}
__device__ __forceinline__ unsigned int pack_bf16x2(float lo, float hi) {
  return (unsigned int)f32_bf16(lo) | ((unsigned int)f32_bf16(hi) << 16);
}

// ---------------------------------------------------------------------------
// Phase 0: convert W (f32, [E][E] row-major) into bf16 packed along K:
//   Wpk[k2*E + n] = pack(W[2*k2][n], W[2*k2+1][n])
// so a WMMA B-fragment dword (two consecutive-K bf16 for one column) is a
// single b32 load.
// ---------------------------------------------------------------------------
__global__ __launch_bounds__(256) void pack_w_kernel(const float* __restrict__ W,
                                                     unsigned int* __restrict__ Wpk) {
  const int idx = blockIdx.x * 256 + threadIdx.x;      // E/2 * E total
  const int n  = idx & (E_ - 1);
  const int k2 = idx >> 10;
  const float* w0 = W + (size_t)(2 * k2) * E_ + n;
  Wpk[idx] = pack_bf16x2(w0[0], w0[E_]);
}

// ---------------------------------------------------------------------------
// Phase 1: Y = GroupNorm(CELU(X @ W + b)) for 32 rows per block.
// 256 threads = 8 waves; wave w computes columns [128w, 128w+128) = group w.
// Register blocking: 2 A-fragments (2 x 16 rows) per wave, each B fragment
// feeds 2 WMMAs -> 16 v_wmma per K-step per wave at the same L2 B-traffic.
// ---------------------------------------------------------------------------
__global__ __launch_bounds__(256) void proj_kernel(
    const float* __restrict__ X, const unsigned int* __restrict__ Wpk,
    const float* __restrict__ bias, const float* __restrict__ gw,
    const float* __restrict__ gb, float* __restrict__ Y)
{
  __shared__ unsigned int Apk[32 * 17];   // 32 rows x 16 K-pair dwords, pad 17

  const int tid  = threadIdx.x;
  const int lane = tid & 31;
  const int wave = tid >> 5;
  const int row0 = blockIdx.x * 32;

  v8f acc[2][8] = {};

  // A-fragment addressing (ISA 16-bit A 16x32 layout):
  //  lanes 0-15 : V0..3 = K{0..7} pairs 0..3 ; V4..7 = K{16..23} pairs 8..11
  //  lanes16-31 : V0..3 = K{8..15} pairs 4..7; V4..7 = K{24..31} pairs 12..15
  const int mA   = lane & 15;
  const int hi   = lane >> 4;
  const int ap1  = hi * 4;
  const int ap2  = 8 + hi * 4;
  // B-fragment: same K order as A (lanes 0-15: K0..7,16..23; lanes 16-31: K8..15,24..31).
  const int nB  = lane & 15;
  const int kB2 = hi * 4;

  for (int kb = 0; kb < E_; kb += 32) {
    // stage 32x32 f32 A tile -> packed bf16 pairs in LDS (2 dwords / thread)
    #pragma unroll
    for (int it = 0; it < 2; ++it) {
      const int idx = tid + it * 256;
      const int m  = idx >> 4;
      const int k2 = idx & 15;
      const float* xp = X + (size_t)(row0 + m) * E_ + kb + 2 * k2;
      Apk[m * 17 + k2] = pack_bf16x2(xp[0], xp[1]);
    }
    __syncthreads();

    Frag16 a[2];
    #pragma unroll
    for (int r = 0; r < 2; ++r) {
      const int base = (r * 16 + mA) * 17;
      #pragma unroll
      for (int jj = 0; jj < 4; ++jj) {
        a[r].d[jj]     = Apk[base + ap1 + jj];
        a[r].d[4 + jj] = Apk[base + ap2 + jj];
      }
    }

    const unsigned int* Wb0 = Wpk + (size_t)((kb >> 1) + kB2) * E_;
    #pragma unroll
    for (int t = 0; t < 8; ++t) {
      const int col = wave * 128 + t * 16 + nB;
      Frag16 bf;
      #pragma unroll
      for (int jj = 0; jj < 8; ++jj) bf.d[jj] = Wb0[(size_t)(jj + (jj >> 2) * 4) * E_ + col];
      acc[0][t] = __builtin_amdgcn_wmma_f32_16x16x32_bf16(
          false, a[0].v, false, bf.v, (short)0, acc[0][t], false, false);
      acc[1][t] = __builtin_amdgcn_wmma_f32_16x16x32_bf16(
          false, a[1].v, false, bf.v, (short)0, acc[1][t], false, false);
    }
    __syncthreads();
  }

  // Epilogue: bias + CELU, then GroupNorm over this wave's 128 columns,
  // independently for each of the 2 row-groups.
  // C/D layout: element j of acc = row (j + 8*hi), col = wave*128 + t*16 + (lane&15)
  float cgw[8], cgb[8], cbias[8];
  #pragma unroll
  for (int t = 0; t < 8; ++t) {
    const int col = wave * 128 + t * 16 + nB;
    cbias[t] = bias[col];
    cgw[t] = gw[col];
    cgb[t] = gb[col];
  }

  #pragma unroll
  for (int r = 0; r < 2; ++r) {
    float val[8][8];
    #pragma unroll
    for (int t = 0; t < 8; ++t) {
      #pragma unroll
      for (int j = 0; j < 8; ++j) {
        float x = acc[r][t][j] + cbias[t];
        x = (x > 0.0f) ? x : ALPHA_ * (__expf(x * (1.0f / ALPHA_)) - 1.0f);
        val[t][j] = x;
      }
    }
    #pragma unroll
    for (int j = 0; j < 8; ++j) {
      float s = 0.0f, q = 0.0f;
      #pragma unroll
      for (int t = 0; t < 8; ++t) { s += val[t][j]; q += val[t][j] * val[t][j]; }
      #pragma unroll
      for (int off = 1; off < 16; off <<= 1) {   // reduce within 16-lane half
        s += __shfl_xor(s, off, 32);
        q += __shfl_xor(q, off, 32);
      }
      const float mu  = s * (1.0f / 128.0f);
      const float var = q * (1.0f / 128.0f) - mu * mu;
      const float rs  = rsqrtf(var + EPS_);
      const int row = row0 + r * 16 + j + hi * 8;
      #pragma unroll
      for (int t = 0; t < 8; ++t) {
        const int col = wave * 128 + t * 16 + nB;
        Y[(size_t)row * E_ + col] = (val[t][j] - mu) * rs * cgw[t] + cgb[t];
      }
    }
  }
}

// ---------------------------------------------------------------------------
// Phase 2: per (b,h) attention block. hidden[196][64] in LDS (~49KB).
// ---------------------------------------------------------------------------
__global__ __launch_bounds__(256) void attn_kernel(
    const float* __restrict__ qp, const float* __restrict__ kp,
    const int* __restrict__ mask, const float* __restrict__ v1p,
    const float* __restrict__ v2p, const float* __restrict__ Wb,
    const float* __restrict__ bb, const float* __restrict__ Ws,
    const float* __restrict__ bs, const float* __restrict__ Wc,
    const float* __restrict__ bc, float* __restrict__ out)
{
  __shared__ float qs[Dh_];
  __shared__ float hid[M_ * MID_];
  __shared__ float alpha[M_];
  __shared__ float pooled[MID_];
  __shared__ float smax_s, ssum_s;

  const int tid = threadIdx.x;
  const int b = blockIdx.x >> 3;
  const int h = blockIdx.x & 7;

  const float* krows  = kp  + (size_t)b * M_ * E_ + h * Dh_;
  const float* v2rows = v2p + (size_t)b * M_ * E_ + h * Dh_;
  const int* mrow = mask + b * M_;

  if (tid < Dh_) qs[tid] = qp[(size_t)b * E_ + h * Dh_ + tid];
  __syncthreads();

  // hidden[m][j] = relu( (q*k[m]) . Wb[:,j] + bb[j] )
  const int j  = tid & 63;
  const int mq = tid >> 6;
  for (int m = mq; m < M_; m += 4) {
    const float* kr = krows + (size_t)m * E_;
    __builtin_prefetch(kr + 4 * E_, 0, 3);
    float a = bb[j];
    #pragma unroll 4
    for (int d = 0; d < Dh_; ++d) a = fmaf(qs[d] * kr[d], Wb[d * MID_ + j], a);
    hid[m * MID_ + j] = fmaxf(a, 0.0f);
  }
  __syncthreads();

  // masked scores
  if (tid < M_) {
    float s = bs[0];
    #pragma unroll 8
    for (int jj = 0; jj < MID_; ++jj) s = fmaf(hid[tid * MID_ + jj], Ws[jj], s);
    alpha[tid] = (mrow[tid] == 0) ? -1.0e9f : s;
  }
  __syncthreads();

  // softmax over M (wave 0 reduces)
  if (tid < 32) {
    float mx = -1.0e30f;
    for (int m = tid; m < M_; m += 32) mx = fmaxf(mx, alpha[m]);
    #pragma unroll
    for (int off = 16; off; off >>= 1) mx = fmaxf(mx, __shfl_xor(mx, off, 32));
    float sm = 0.0f;
    for (int m = tid; m < M_; m += 32) sm += __expf(alpha[m] - mx);
    #pragma unroll
    for (int off = 16; off; off >>= 1) sm += __shfl_xor(sm, off, 32);
    if (tid == 0) { smax_s = mx; ssum_s = sm; }
  }
  __syncthreads();
  if (tid < M_) alpha[tid] = __expf(alpha[tid] - smax_s) / ssum_s;

  // masked mean-pool of hidden over M
  if (tid < MID_) {
    float p = 0.0f, cnt = 0.0f;
    for (int m = 0; m < M_; ++m) {
      const float mk = (float)mrow[m];
      p = fmaf(hid[m * MID_ + tid], mk, p);
      cnt += mk;
    }
    pooled[tid] = p / cnt;
  }
  __syncthreads();

  if (tid < Dh_) {
    const int d = tid;
    float ac = bc[d];
    #pragma unroll 8
    for (int jj = 0; jj < MID_; ++jj) ac = fmaf(pooled[jj], Wc[jj * Dh_ + d], ac);
    ac = 1.0f / (1.0f + __expf(-ac));
    float vp = 0.0f;
    for (int m = 0; m < M_; ++m) vp = fmaf(alpha[m], v2rows[(size_t)m * E_ + d], vp);
    const float v1v = v1p[(size_t)b * E_ + h * Dh_ + d];
    out[(size_t)b * E_ + h * Dh_ + d] = v1v * vp * ac;
  }
}

// ---------------------------------------------------------------------------
extern "C" void kernel_launch(void* const* d_in, const int* in_sizes, int n_in,
                              void* d_out, int out_size, void* d_ws, size_t ws_size,
                              hipStream_t stream) {
  const float* query  = (const float*)d_in[0];
  const float* key    = (const float*)d_in[1];
  const int*   mask   = (const int*)  d_in[2];
  const float* value1 = (const float*)d_in[3];
  const float* value2 = (const float*)d_in[4];
  const float* Wq = (const float*)d_in[5],  *bq = (const float*)d_in[6],
             *gqw = (const float*)d_in[7],  *gqb = (const float*)d_in[8];
  const float* Wk = (const float*)d_in[9],  *bk = (const float*)d_in[10],
             *gkw = (const float*)d_in[11], *gkb = (const float*)d_in[12];
  const float* Wv1 = (const float*)d_in[13], *bv1 = (const float*)d_in[14],
             *gv1w = (const float*)d_in[15], *gv1b = (const float*)d_in[16];
  const float* Wv2 = (const float*)d_in[17], *bv2 = (const float*)d_in[18],
             *gv2w = (const float*)d_in[19], *gv2b = (const float*)d_in[20];
  const float* Wb = (const float*)d_in[21], *bb = (const float*)d_in[22];
  const float* Ws = (const float*)d_in[23], *bs = (const float*)d_in[24];
  const float* Wc = (const float*)d_in[25], *bc = (const float*)d_in[26];

  // Workspace partition (needs ~421 MB): 4x packed-bf16 weights (2MB each),
  // then f32 projected q, v1, k, v2.
  unsigned char* ws = (unsigned char*)d_ws;
  const size_t wpkB = (size_t)(E_ / 2) * E_ * sizeof(unsigned int);
  unsigned int* WqP  = (unsigned int*)(ws + 0 * wpkB);
  unsigned int* WkP  = (unsigned int*)(ws + 1 * wpkB);
  unsigned int* Wv1P = (unsigned int*)(ws + 2 * wpkB);
  unsigned int* Wv2P = (unsigned int*)(ws + 3 * wpkB);
  float* qpj  = (float*)(ws + 4 * wpkB);
  float* v1pj = qpj  + (size_t)B_ * E_;
  float* kpj  = v1pj + (size_t)B_ * E_;
  float* v2pj = kpj  + (size_t)B_ * M_ * E_;

  const dim3 blk(256);
  const int packGrid = (E_ / 2) * E_ / 256;
  pack_w_kernel<<<packGrid, blk, 0, stream>>>(Wq,  WqP);
  pack_w_kernel<<<packGrid, blk, 0, stream>>>(Wk,  WkP);
  pack_w_kernel<<<packGrid, blk, 0, stream>>>(Wv1, Wv1P);
  pack_w_kernel<<<packGrid, blk, 0, stream>>>(Wv2, Wv2P);

  proj_kernel<<<B_ / 32,        blk, 0, stream>>>(query,  WqP,  bq,  gqw,  gqb,  qpj);
  proj_kernel<<<B_ / 32,        blk, 0, stream>>>(value1, Wv1P, bv1, gv1w, gv1b, v1pj);
  proj_kernel<<<(B_ * M_) / 32, blk, 0, stream>>>(key,    WkP,  bk,  gkw,  gkb,  kpj);
  proj_kernel<<<(B_ * M_) / 32, blk, 0, stream>>>(value2, Wv2P, bv2, gv2w, gv2b, v2pj);

  attn_kernel<<<B_ * H_, blk, 0, stream>>>(qpj, kpj, mask, v1pj, v2pj,
                                           Wb, bb, Ws, bs, Wc, bc, (float*)d_out);
}
